// MMGroupAttention_14010183320092
// MI455X (gfx1250) — matrix operand order repaired and checked by reading, not compiled, not run on hardware
//
#include <hip/hip_runtime.h>
#include <hip/hip_bf16.h>

// ---------------------------------------------------------------------------
// Problem constants (from reference): B=8, S=1024, D=1024, H=8, DK=128
// ---------------------------------------------------------------------------
#define Bn 8
#define Sn 1024
#define Dn 1024
#define Hn 8
#define DKn 128

typedef __attribute__((ext_vector_type(8)))  unsigned short v8us;
typedef __attribute__((ext_vector_type(16))) unsigned short v16us;
typedef __attribute__((ext_vector_type(16))) __bf16        v16bf;
typedef __attribute__((ext_vector_type(8)))  float         v8f;

// float -> bf16 round-to-nearest-even
__device__ __forceinline__ unsigned short f2bf(float f) {
    unsigned int u = __float_as_uint(f);
    u += 0x7FFFu + ((u >> 16) & 1u);
    return (unsigned short)(u >> 16);
}

// A-fragment (16x32 bf16, MxK): lane L holds row L&15; K = (L>>4)*16 .. +15,
// contiguous, in elements 0..15.
__device__ __forceinline__ v16bf load_a_frag(const unsigned short* p, int ld, int lane) {
    const int row = lane & 15;
    const int k0  = (lane >> 4) << 4;          // 0 or 16
    union AU { struct { v8us lo, hi; } s; v16bf v; } u;
    u.s.lo = *(const v8us*)(p + row * ld + k0);
    u.s.hi = *(const v8us*)(p + row * ld + k0 + 8);
    return u.v;
}

// B-fragment (32x16 bf16, KxN) sourced from a row-major [N][K] array (i.e. B^T):
// lane L holds column N = L&15; K = (L>>4)*16 .. +15, contiguous.
__device__ __forceinline__ v16bf load_b_frag(const unsigned short* p, int ld, int lane) {
    const int col = lane & 15;
    const int k0  = (lane >> 4) << 4;          // 0 or 16
    union BU { v16us a; v16bf v; } u;
    u.a = *(const v16us*)(p + col * ld + k0);
    return u.v;
}

__device__ __forceinline__ v8f wmma_bf16(v16bf a, v16bf b, v8f c) {
    return __builtin_amdgcn_wmma_f32_16x16x32_bf16(
        /*neg_a=*/false, a, /*neg_b=*/false, b,
        /*c_mod=*/(short)0, c, /*reuse_a=*/false, /*reuse_b=*/false);
}

// CDNA5 async global->LDS copy (16B), tracked by ASYNCcnt (ISA 15.18.3 op 98).
// VDST VGPR holds the LDS byte address (low 32 bits of the generic pointer).
__device__ __forceinline__ void async_b128(void* lds_dst, const void* gsrc) {
    unsigned          loff = (unsigned)(unsigned long long)lds_dst;
    unsigned long long ga  = (unsigned long long)gsrc;
    asm volatile("global_load_async_to_lds_b128 %0, %1, off"
                 :: "v"(loff), "v"(ga) : "memory");
}
__device__ __forceinline__ void wait_async0() {
    asm volatile("s_wait_asynccnt 0x0" ::: "memory");
}

// reductions across the 16-lane half (xor masks < 16 stay within each half)
__device__ __forceinline__ float rowmax16(float v) {
    #pragma unroll
    for (int m = 8; m >= 1; m >>= 1) v = fmaxf(v, __shfl_xor(v, m, 32));
    return v;
}
__device__ __forceinline__ float rowsum16(float v) {
    #pragma unroll
    for (int m = 8; m >= 1; m >>= 1) v += __shfl_xor(v, m, 32);
    return v;
}

// ---------------------------------------------------------------------------
// fp32 -> bf16 conversion (vectorized, n % 4 == 0 for all our tensors)
// ---------------------------------------------------------------------------
__global__ __launch_bounds__(256) void cvt_f32_bf16(const float* __restrict__ in,
                                                    unsigned short* __restrict__ out,
                                                    int n) {
    int i = (blockIdx.x * blockDim.x + threadIdx.x) * 4;
    if (i < n) {
        float4 f = *(const float4*)(in + i);
        union { unsigned short u[4]; uint2 v; } o;
        o.u[0] = f2bf(f.x); o.u[1] = f2bf(f.y);
        o.u[2] = f2bf(f.z); o.u[3] = f2bf(f.w);
        *(uint2*)(out + i) = o.v;
    }
}

// ---------------------------------------------------------------------------
// WMMA GEMM: Y[M,N] = A[M,K] @ W[N,K]^T + bias   (torch Linear convention)
// Block = 8 waves; block tile = 128x64. Both the 128x32 A tile and the 64x32
// B tile are staged in LDS (double-buffered) with async-to-LDS DMA; the
// steady-state loop touches global memory only through ASYNCcnt-tracked DMA.
// mode 0: bf16 out row-major [M,N]; mode 1: bf16 scatter to Vt[B,H,DK,S];
// mode 2: fp32 out row-major [M,N]
// ---------------------------------------------------------------------------
__global__ __launch_bounds__(256) void gemm_wmma(const unsigned short* __restrict__ A,
                                                 const unsigned short* __restrict__ W,
                                                 const float* __restrict__ bias,
                                                 unsigned short* __restrict__ outb,
                                                 float* __restrict__ outf,
                                                 int M, int N, int K, int mode) {
    __shared__ unsigned short ldsA[2][128 * 32];   // 2 x 8KB
    __shared__ unsigned short ldsB[2][64 * 32];    // 2 x 4KB
    const int tid  = threadIdx.x;
    const int lane = tid & 31;
    const int w    = tid >> 5;
    const int ntg  = N >> 6;                       // 64-wide N groups
    const int bm   = blockIdx.x / ntg;             // 128-row block index
    const int bn   = blockIdx.x % ntg;
    const int m0   = (bm * 8 + w) * 16;
    const int n0   = bn * 64;

    // staging coordinates: B 64x32 (1 x b128/thread), A 128x32 (1 x b128/thread)
    const int brow = tid >> 2, bc8 = (tid & 3) * 8;      // 64 rows x 32
    const int arow = tid >> 1, ac16 = (tid & 1) * 16;    // 128 rows x 32
    const unsigned short* wtile = W + (size_t)(n0 + brow) * K + bc8;
    const unsigned short* atile = A + (size_t)(bm * 128 + arow) * K + ac16;

    // prologue: stage k-chunk 0
    async_b128(&ldsB[0][brow * 32 + bc8], wtile);
    async_b128(&ldsA[0][arow * 32 + ac16], atile);
    wait_async0();
    __syncthreads();

    v8f acc[4] = {};
    const int kiters = K >> 5;
    int buf = 0;
    for (int i = 0; i < kiters - 1; ++i) {
        const int kn = (i + 1) << 5;
        // issue next chunk's DMA into the other buffer
        async_b128(&ldsB[buf ^ 1][brow * 32 + bc8], wtile + kn);
        async_b128(&ldsA[buf ^ 1][arow * 32 + ac16], atile + kn);
        // compute from current buffer: batch all fragment loads, then WMMA
        v16bf afr = load_a_frag(&ldsA[buf][w * 16 * 32], 32, lane);
        v16bf bfr[4];
        #pragma unroll
        for (int t = 0; t < 4; ++t)
            bfr[t] = load_b_frag(&ldsB[buf][t * 16 * 32], 32, lane);
        #pragma unroll
        for (int t = 0; t < 4; ++t)
            acc[t] = wmma_bf16(afr, bfr[t], acc[t]);
        wait_async0();
        __syncthreads();
        buf ^= 1;
    }
    {   // peeled last chunk: no DMA issue, no barrier
        v16bf afr = load_a_frag(&ldsA[buf][w * 16 * 32], 32, lane);
        v16bf bfr[4];
        #pragma unroll
        for (int t = 0; t < 4; ++t)
            bfr[t] = load_b_frag(&ldsB[buf][t * 16 * 32], 32, lane);
        #pragma unroll
        for (int t = 0; t < 4; ++t)
            acc[t] = wmma_bf16(afr, bfr[t], acc[t]);
    }

    const int col  = lane & 15;
    const int rofs = (lane >> 4) << 3;             // C row = rofs + r
    #pragma unroll
    for (int t = 0; t < 4; ++t) {
        const int n  = n0 + t * 16 + col;
        const float bv = bias ? bias[n] : 0.f;
        #pragma unroll
        for (int r = 0; r < 8; ++r) {
            const int m = m0 + rofs + r;
            const float y = acc[t][r] + bv;
            if (mode == 0) {
                outb[(size_t)m * N + n] = f2bf(y);
            } else if (mode == 2) {
                outf[(size_t)m * N + n] = y;
            } else {                               // Vt[B,H,DK,S] scatter
                const int bidx = m >> 10, s = m & (Sn - 1);
                const int h = n >> 7, dk = n & (DKn - 1);
                outb[(((size_t)bidx * Hn + h) * DKn + dk) * Sn + s] = f2bf(y);
            }
        }
    }
}

// ---------------------------------------------------------------------------
// Flash attention: block = 8 waves = 8 consecutive 16-row q tiles of one
// (b,h). K (32x128) and Vt (128x32) chunk tiles are staged in LDS with async
// DMA (double-buffered) and shared by all 8 waves. Per 32-key chunk:
// 8 WMMA for Q K^T, masked online softmax, P staged via per-wave LDS into
// A-fragment layout, 8 WMMA for P V.
// ---------------------------------------------------------------------------
__global__ __launch_bounds__(256) void attn_wmma(const unsigned short* __restrict__ Qp,
                                                 const unsigned short* __restrict__ Kp,
                                                 const unsigned short* __restrict__ Vt,
                                                 const int* __restrict__ mask,
                                                 unsigned short* __restrict__ X) {
    __shared__ unsigned short ldsK[2][32 * 128];   // 2 x 8KB
    __shared__ unsigned short ldsV[2][128 * 32];   // 2 x 8KB
    __shared__ unsigned short ldsP[8][512];        // 16x32 bf16 per wave
    const int tid  = threadIdx.x;
    const int lane = tid & 31;
    const int w    = tid >> 5;
    const int wid  = blockIdx.x * 8 + w;
    const int b  = wid >> 9;                       // / (H * S/16)
    const int h  = (wid >> 6) & 7;
    const int qt = wid & 63;

    const unsigned short* qbase = Qp + ((size_t)(b * Sn + qt * 16)) * Dn + h * DKn;
    const unsigned short* kbase = Kp + (size_t)b * Sn * Dn + h * DKn;
    const unsigned short* vbase = Vt + ((size_t)(b * Hn + h)) * DKn * Sn;
    const int* mbase = mask + ((size_t)(b * 4 + (h & 3)) * Sn + qt * 16) * Sn;

    // staging coordinates (per thread, 1 x 16B for K, 1 x 16B for V per chunk)
    const int krow = tid >> 3, kseg = (tid & 7) * 16;    // 32 x 128
    const int vrow = tid >> 1, vseg = (tid & 1) * 16;    // 128 x 32

    v16bf qf[4];
    #pragma unroll
    for (int c = 0; c < 4; ++c) qf[c] = load_a_frag(qbase + c * 32, Dn, lane);

    // prologue: stage chunk 0
    async_b128(&ldsK[0][krow * 128 + kseg], kbase + (size_t)krow * Dn + kseg);
    async_b128(&ldsV[0][vrow * 32 + vseg], vbase + (size_t)vrow * Sn + vseg);
    wait_async0();
    __syncthreads();

    v8f acc[8] = {};
    float mrun[8], lrun[8];
    #pragma unroll
    for (int r = 0; r < 8; ++r) { mrun[r] = -3.0e38f; lrun[r] = 0.f; }

    const int col  = lane & 15;
    const int rofs = (lane >> 4) << 3;
    const float scale = 0.088388347648318447f;     // 1/sqrt(128)
    unsigned short* lp = &ldsP[w][0];

    // one 32-key chunk of work against LDS buffer `cb`
    auto chunk = [&](int jc, int cb) {
        const int k32 = jc * 32;
        // ---- scores: Q (16x128) x K^T (128x32) -> two 16x16 tiles ----
        v8f s0 = {}, s1 = {};
        #pragma unroll
        for (int c = 0; c < 4; ++c) {
            v16bf kf0 = load_b_frag(&ldsK[cb][0 * 128 + c * 32], 128, lane);
            v16bf kf1 = load_b_frag(&ldsK[cb][16 * 128 + c * 32], 128, lane);
            s0 = wmma_bf16(qf[c], kf0, s0);
            s1 = wmma_bf16(qf[c], kf1, s1);
        }
        // ---- scale + mask ----
        float p0[8], p1[8];
        #pragma unroll
        for (int r = 0; r < 8; ++r) {
            const int* mrow = mbase + (size_t)(rofs + r) * Sn + k32;
            float x0 = s0[r] * scale;
            float x1 = s1[r] * scale;
            if (mrow[col] == 0)      x0 = -1e9f;
            if (mrow[col + 16] == 0) x1 = -1e9f;
            p0[r] = x0; p1[r] = x1;
        }
        // ---- online softmax update ----
        #pragma unroll
        for (int r = 0; r < 8; ++r) {
            const float rm = rowmax16(fmaxf(p0[r], p1[r]));
            const float nm = fmaxf(mrun[r], rm);
            const float alpha = __expf(mrun[r] - nm);
            mrun[r] = nm;
            const float e0 = __expf(p0[r] - nm);
            const float e1 = __expf(p1[r] - nm);
            lrun[r] = lrun[r] * alpha + rowsum16(e0 + e1);
            p0[r] = e0; p1[r] = e1;
            #pragma unroll
            for (int t = 0; t < 8; ++t) acc[t][r] *= alpha;
        }
        // ---- stage P (C layout) through LDS into A-fragment layout ----
        #pragma unroll
        for (int r = 0; r < 8; ++r) {
            const int row = rofs + r;
            lp[row * 32 + col]      = f2bf(p0[r]);
            lp[row * 32 + col + 16] = f2bf(p1[r]);
        }
        asm volatile("s_wait_dscnt 0x0" ::: "memory");
        v16bf pf = load_a_frag(lp, 32, lane);
        // ---- out += P (16x32) x V (32x128), V tile from LDS ----
        #pragma unroll
        for (int t = 0; t < 8; ++t) {
            v16bf vf = load_b_frag(&ldsV[cb][t * 16 * 32], 32, lane);
            acc[t] = wmma_bf16(pf, vf, acc[t]);
        }
    };

    int buf = 0;
    const int nchunks = Sn / 32;
    for (int jc = 0; jc < nchunks - 1; ++jc) {
        const int kn = (jc + 1) * 32;
        async_b128(&ldsK[buf ^ 1][krow * 128 + kseg],
                   kbase + (size_t)(kn + krow) * Dn + kseg);
        async_b128(&ldsV[buf ^ 1][vrow * 32 + vseg],
                   vbase + (size_t)vrow * Sn + kn + vseg);
        chunk(jc, buf);
        wait_async0();
        __syncthreads();
        buf ^= 1;
    }
    chunk(nchunks - 1, buf);                       // peeled last chunk

    // ---- epilogue: divide by softmax denom, write x[b,s,h*DK+dk] ----
    float inv[8];
    #pragma unroll
    for (int r = 0; r < 8; ++r) inv[r] = 1.f / lrun[r];
    unsigned short* xb = X + ((size_t)(b * Sn + qt * 16)) * Dn + h * DKn;
    #pragma unroll
    for (int t = 0; t < 8; ++t)
        #pragma unroll
        for (int r = 0; r < 8; ++r)
            xb[(size_t)(rofs + r) * Dn + t * 16 + col] = f2bf(acc[t][r] * inv[r]);
}

// ---------------------------------------------------------------------------
// Host launcher
// ---------------------------------------------------------------------------
extern "C" void kernel_launch(void* const* d_in, const int* in_sizes, int n_in,
                              void* d_out, int out_size, void* d_ws, size_t ws_size,
                              hipStream_t stream) {
    (void)in_sizes; (void)n_in; (void)out_size; (void)ws_size;

    const float* q    = (const float*)d_in[0];
    const float* k    = (const float*)d_in[1];
    const float* v    = (const float*)d_in[2];
    const int*   mask = (const int*)d_in[3];
    const float* Wq   = (const float*)d_in[4];
    const float* bq   = (const float*)d_in[5];
    const float* Wk   = (const float*)d_in[6];
    const float* bk   = (const float*)d_in[7];
    const float* Wv   = (const float*)d_in[8];
    const float* bv   = (const float*)d_in[9];

    const size_t SBD = (size_t)Bn * Sn * Dn;   // 8,388,608 elements
    const size_t DD  = (size_t)Dn * Dn;        // 1,048,576 elements

    unsigned short* qb  = (unsigned short*)d_ws;
    unsigned short* kb  = qb  + SBD;
    unsigned short* vb  = kb  + SBD;
    unsigned short* wqb = vb  + SBD;
    unsigned short* wkb = wqb + DD;
    unsigned short* wvb = wkb + DD;
    unsigned short* Qp  = wvb + DD;
    unsigned short* Kp  = Qp  + SBD;
    unsigned short* Vt  = Kp  + SBD;
    unsigned short* Xb  = Vt  + SBD;           // total ~123.6 MB of workspace

    // 1) fp32 -> bf16 conversions
    cvt_f32_bf16<<<(int)(SBD / 1024), 256, 0, stream>>>(q,  qb,  (int)SBD);
    cvt_f32_bf16<<<(int)(SBD / 1024), 256, 0, stream>>>(k,  kb,  (int)SBD);
    cvt_f32_bf16<<<(int)(SBD / 1024), 256, 0, stream>>>(v,  vb,  (int)SBD);
    cvt_f32_bf16<<<(int)(DD  / 1024), 256, 0, stream>>>(Wq, wqb, (int)DD);
    cvt_f32_bf16<<<(int)(DD  / 1024), 256, 0, stream>>>(Wk, wkb, (int)DD);
    cvt_f32_bf16<<<(int)(DD  / 1024), 256, 0, stream>>>(Wv, wvb, (int)DD);

    // 2) projections: blocks = (M/128)*(N/64) = 64*16 = 1024, 8 waves each
    const int M = Bn * Sn, N = Dn, K = Dn;
    const int gemm_blocks = (M >> 7) * (N >> 6);
    gemm_wmma<<<gemm_blocks, 256, 0, stream>>>(qb, wqb, bq, Qp, nullptr, M, N, K, 0);
    gemm_wmma<<<gemm_blocks, 256, 0, stream>>>(kb, wkb, bk, Kp, nullptr, M, N, K, 0);
    gemm_wmma<<<gemm_blocks, 256, 0, stream>>>(vb, wvb, bv, Vt, nullptr, M, N, K, 1);

    // 3) flash attention: 4096 waves -> 512 blocks
    attn_wmma<<<(Bn * Hn * (Sn / 16)) / 8, 256, 0, stream>>>(Qp, Kp, Vt, mask, Xb);

    // 4) output projection (reference reuses Wv/bv), fp32 out
    gemm_wmma<<<gemm_blocks, 256, 0, stream>>>(Xb, wvb, bv, nullptr, (float*)d_out,
                                               M, N, K, 2);
}
